// AxialAttention_75806172774610
// MI455X (gfx1250) — compile-verified, matrix operand order checked by reading, not run on hardware
//
#include <hip/hip_runtime.h>
#include <hip/hip_bf16.h>

#define CC    256
#define HEADS 8
#define DIM   32
#define LSEQ  128      // attention length (= H)
#define WW    128      // W
#define BT    16       // B*T
#define NSEQ  (BT*WW)  // 2048 sequences
#define HW    (LSEQ*WW)
#define SCALE 0.17677669529663687f  // 1/sqrt(32)

typedef __attribute__((ext_vector_type(16))) _Float16 v16h;
typedef __attribute__((ext_vector_type(8)))  _Float16 v8h;
typedef __attribute__((ext_vector_type(8)))  float    v8f;
typedef _Float16 f16;

union F16x16 { v16h v; v8h h8[2]; };
union F16x2  { f16 h2[2]; unsigned u; };

// ---------------- weight fp32 -> fp16 conversion ----------------
__global__ void wconv_kernel(const float* __restrict__ wq, const float* __restrict__ wp,
                             f16* __restrict__ wq_h, f16* __restrict__ wp_h) {
    int i = blockIdx.x * 256 + threadIdx.x;
    if (i < 3 * CC * CC) wq_h[i] = (f16)wq[i];
    if (i < CC * CC)     wp_h[i] = (f16)wp[i];
}

// ---------------- pass 1: QKV GEMM per (bt,h) slab ----------------
// M=w(128), N=768(qkv ch), K=c(256). A = x[b,t,:,h,:]^T staged transposed in LDS.
#define KP 264   // LDS row stride in f16: 528B = 33*16 (16B aligned rows), 132 dwords
__global__ __launch_bounds__(256, 2) void qkv_kernel(
    const float* __restrict__ x, const f16* __restrict__ wqkv,
    const float* __restrict__ bqkv,
    f16* __restrict__ q_ws, f16* __restrict__ k_ws, f16* __restrict__ v_ws) {
    __shared__ f16 Alds[128 * KP];
    const int blk = blockIdx.x;
    const int bt  = blk >> 7;
    const int h   = blk & 127;
    const float* xs = x + ((size_t)bt * CC) * HW + (size_t)h * WW;

    const int tid = threadIdx.x;
    // stage + transpose + f16-convert: coalesced along w, paired-c b32 LDS stores
    {
        const int w  = tid & 127;
        const int ch = (tid >> 7) * 128;
        for (int cp = 0; cp < 64; ++cp) {
            const int c = ch + cp * 2;
            F16x2 pk;
            pk.h2[0] = (f16)xs[(size_t)c * HW + w];
            pk.h2[1] = (f16)xs[(size_t)(c + 1) * HW + w];
            *(unsigned*)&Alds[w * KP + c] = pk.u;
        }
    }
    __syncthreads();

    const int wave = tid >> 5, lane = tid & 31;
    const int lhalf = lane >> 4, l15 = lane & 15;
    const int nbase = wave * 96;  // 6 N-tiles per wave

    for (int mt = 0; mt < 8; ++mt) {
        v8f acc[6] = {};
        const int m = mt * 16 + l15;
        for (int kt = 0; kt < 8; ++kt) {
            const int k0 = kt * 32 + lhalf * 8;
            F16x16 a;
            a.h8[0] = *(const v8h*)&Alds[m * KP + k0];
            a.h8[1] = *(const v8h*)&Alds[m * KP + k0 + 16];
            for (int jt = 0; jt < 6; ++jt) {
                const int j = nbase + jt * 16 + l15;
                const f16* bp = wqkv + (size_t)j * CC + k0;
                F16x16 b;
                b.h8[0] = *(const v8h*)bp;
                b.h8[1] = *(const v8h*)(bp + 16);
                acc[jt] = __builtin_amdgcn_wmma_f32_16x16x32_f16(
                    false, a.v, false, b.v, (short)0, acc[jt], false, false);
            }
        }
        // epilogue: +bias, q-prescale, scatter f16 to [n][head][l][d]
        for (int jt = 0; jt < 6; ++jt) {
            const int j    = nbase + jt * 16 + l15;
            const int sel  = j >> 8;      // 0=q 1=k 2=v (tiles never straddle)
            const int jj   = j & 255;
            const int head = jj >> 5;
            const int d    = jj & 31;
            const float bv = bqkv[j];
            const float mul = (sel == 0) ? SCALE : 1.0f;
            f16* dst = (sel == 0) ? q_ws : ((sel == 1) ? k_ws : v_ws);
            for (int i = 0; i < 8; ++i) {
                const int wr = mt * 16 + i + lhalf * 8;   // w index
                const float val = (acc[jt][i] + bv) * mul;
                const size_t idx =
                    ((((size_t)(bt * WW + wr)) * HEADS + head) * LSEQ + h) * DIM + d;
                dst[idx] = (f16)val;
            }
        }
    }
}

// ---------------- pass 2: attention per (n,head) ----------------
#define KLP 40    // K rows: 80B stride (16B aligned, conflict-free frags)
#define VTP 136   // V^T rows: 272B stride
#define PLP 136   // per-wave P rows: 272B stride
__global__ __launch_bounds__(128, 2) void attn_kernel(
    const f16* __restrict__ q_ws, const f16* __restrict__ k_ws,
    const f16* __restrict__ v_ws, const float* __restrict__ rbias,
    f16* __restrict__ o_ws) {
    __shared__ f16 Klds[128 * KLP];
    __shared__ f16 Vt[DIM * VTP];
    __shared__ f16 Plds[4 * 32 * PLP];

    const int blk  = blockIdx.x;
    const int n    = blk >> 3;
    const int head = blk & 7;
    const f16* qb = q_ws + ((size_t)n * HEADS + head) * LSEQ * DIM;
    const f16* kb = k_ws + ((size_t)n * HEADS + head) * LSEQ * DIM;
    const f16* vb = v_ws + ((size_t)n * HEADS + head) * LSEQ * DIM;
    const float* bb = rbias + (size_t)head * LSEQ * LSEQ;

    const int tid = threadIdx.x;
    // stage K naturally: thread t -> row t (coalesced 64B per lane)
    {
        const v8h* src = (const v8h*)(kb + (size_t)tid * DIM);
        v8h* dst = (v8h*)&Klds[tid * KLP];
        dst[0] = src[0]; dst[1] = src[1]; dst[2] = src[2]; dst[3] = src[3];
    }
    // stage V transposed: Vt[d][l]
    {
        const int d = tid & 31;
        const int lbase = (tid >> 5) * 32;
        for (int i = 0; i < 32; ++i) {
            const int l = lbase + i;
            Vt[d * VTP + l] = vb[(size_t)l * DIM + d];
        }
    }
    __syncthreads();

    const int wave = tid >> 5, lane = tid & 31;
    const int lhalf = lane >> 4, l15 = lane & 15;
    const int mbase = wave * 32;
    f16* Pw = &Plds[wave * 32 * PLP];

    // S = q k^T (+bias), softmax -> P (f16 in per-wave LDS)
    for (int mt = 0; mt < 2; ++mt) {
        const int m0 = mbase + mt * 16;
        F16x16 a;
        {
            const f16* qr = qb + (size_t)(m0 + l15) * DIM + lhalf * 8;
            a.h8[0] = *(const v8h*)qr;
            a.h8[1] = *(const v8h*)(qr + 16);
        }
        v8f s[8] = {};
        for (int nt = 0; nt < 8; ++nt) {
            const f16* kr = &Klds[(nt * 16 + l15) * KLP + lhalf * 8];
            F16x16 b;
            b.h8[0] = *(const v8h*)kr;
            b.h8[1] = *(const v8h*)(kr + 16);
            s[nt] = __builtin_amdgcn_wmma_f32_16x16x32_f16(
                false, a.v, false, b.v, (short)0, s[nt], false, false);
        }
        for (int i = 0; i < 8; ++i) {
            const int l = m0 + i + lhalf * 8;
            float e[8];
            float mx = -3.0e38f;
            for (int nt = 0; nt < 8; ++nt) {
                const float t = s[nt][i] + bb[l * LSEQ + nt * 16 + l15];
                e[nt] = t;
                mx = fmaxf(mx, t);
            }
            mx = fmaxf(mx, __shfl_xor(mx, 1, 32));
            mx = fmaxf(mx, __shfl_xor(mx, 2, 32));
            mx = fmaxf(mx, __shfl_xor(mx, 4, 32));
            mx = fmaxf(mx, __shfl_xor(mx, 8, 32));
            float sum = 0.f;
            for (int nt = 0; nt < 8; ++nt) { e[nt] = __expf(e[nt] - mx); sum += e[nt]; }
            sum += __shfl_xor(sum, 1, 32);
            sum += __shfl_xor(sum, 2, 32);
            sum += __shfl_xor(sum, 4, 32);
            sum += __shfl_xor(sum, 8, 32);
            const float inv = 1.0f / sum;
            const int prow = mt * 16 + i + lhalf * 8;
            for (int nt = 0; nt < 8; ++nt)
                Pw[prow * PLP + nt * 16 + l15] = (f16)(e[nt] * inv);
        }
    }
    // O = P V  (per-wave rows; P re-read in A-frag layout)
    for (int mt = 0; mt < 2; ++mt) {
        v8f o0 = {}, o1 = {};
        for (int kt = 0; kt < 4; ++kt) {
            const f16* pr = &Pw[(mt * 16 + l15) * PLP + kt * 32 + lhalf * 8];
            F16x16 a;
            a.h8[0] = *(const v8h*)pr;
            a.h8[1] = *(const v8h*)(pr + 16);
            {
                const f16* vr = &Vt[(l15) * VTP + kt * 32 + lhalf * 8];
                F16x16 b;
                b.h8[0] = *(const v8h*)vr;
                b.h8[1] = *(const v8h*)(vr + 16);
                o0 = __builtin_amdgcn_wmma_f32_16x16x32_f16(
                    false, a.v, false, b.v, (short)0, o0, false, false);
            }
            {
                const f16* vr = &Vt[(16 + l15) * VTP + kt * 32 + lhalf * 8];
                F16x16 b;
                b.h8[0] = *(const v8h*)vr;
                b.h8[1] = *(const v8h*)(vr + 16);
                o1 = __builtin_amdgcn_wmma_f32_16x16x32_f16(
                    false, a.v, false, b.v, (short)0, o1, false, false);
            }
        }
        for (int i = 0; i < 8; ++i) {
            const int l = mbase + mt * 16 + i + lhalf * 8;
            const size_t base = ((size_t)n * LSEQ + l) * CC + head * DIM;
            o_ws[base + l15]      = (f16)o0[i];
            o_ws[base + 16 + l15] = (f16)o1[i];
        }
    }
}

// ---------------- pass 3: proj per (bt,h): M=j(256) N=w(128) K=c(256) ----------------
__global__ __launch_bounds__(256, 2) void proj_kernel(
    const f16* __restrict__ o_ws, const f16* __restrict__ wproj,
    const float* __restrict__ bproj, float* __restrict__ out) {
    const int blk = blockIdx.x;
    const int bt = blk >> 7, h = blk & 127;
    const int tid = threadIdx.x;
    const int wave = tid >> 5, lane = tid & 31;
    const int lhalf = lane >> 4, l15 = lane & 15;
    const int jbase = wave * 32;

    for (int mt = 0; mt < 2; ++mt) {
        const int j_arow = jbase + mt * 16 + l15;
        v8f acc[8] = {};
        for (int kt = 0; kt < 8; ++kt) {
            const int k0 = kt * 32 + lhalf * 8;
            const f16* ar = wproj + (size_t)j_arow * CC + k0;
            F16x16 a;
            a.h8[0] = *(const v8h*)ar;
            a.h8[1] = *(const v8h*)(ar + 16);
            for (int nt = 0; nt < 8; ++nt) {
                const int w = nt * 16 + l15;
                const f16* br = o_ws + (((size_t)(bt * WW + w)) * LSEQ + h) * CC + k0;
                F16x16 b;
                b.h8[0] = *(const v8h*)br;
                b.h8[1] = *(const v8h*)(br + 16);
                acc[nt] = __builtin_amdgcn_wmma_f32_16x16x32_f16(
                    false, a.v, false, b.v, (short)0, acc[nt], false, false);
            }
        }
        for (int nt = 0; nt < 8; ++nt) {
            const int w = nt * 16 + l15;
            for (int i = 0; i < 8; ++i) {
                const int j = jbase + mt * 16 + i + lhalf * 8;
                out[(((size_t)bt * CC + j) * LSEQ + h) * WW + w] = acc[nt][i] + bproj[j];
            }
        }
    }
}

extern "C" void kernel_launch(void* const* d_in, const int* in_sizes, int n_in,
                              void* d_out, int out_size, void* d_ws, size_t ws_size,
                              hipStream_t stream) {
    (void)in_sizes; (void)n_in; (void)out_size; (void)ws_size;
    const float* x      = (const float*)d_in[0];
    const float* rbias  = (const float*)d_in[1];
    const float* w_qkv  = (const float*)d_in[2];
    const float* b_qkv  = (const float*)d_in[3];
    const float* w_proj = (const float*)d_in[4];
    const float* b_proj = (const float*)d_in[5];
    float* out = (float*)d_out;

    char* ws = (char*)d_ws;
    const size_t QS = (size_t)NSEQ * HEADS * LSEQ * DIM * sizeof(f16);  // 128 MB
    f16* q_ws = (f16*)(ws);
    f16* k_ws = (f16*)(ws + QS);
    f16* v_ws = (f16*)(ws + 2 * QS);
    f16* o_ws = (f16*)(ws + 3 * QS);                                    // NSEQ*LSEQ*CC == QS
    f16* wq_h = (f16*)(ws + 4 * QS);
    f16* wp_h = (f16*)(ws + 4 * QS + (size_t)3 * CC * CC * sizeof(f16));

    wconv_kernel<<<(3 * CC * CC + 255) / 256, 256, 0, stream>>>(w_qkv, w_proj, wq_h, wp_h);
    qkv_kernel<<<BT * LSEQ, 256, 0, stream>>>(x, wq_h, b_qkv, q_ws, k_ws, v_ws);
    attn_kernel<<<NSEQ * HEADS, 128, 0, stream>>>(q_ws, k_ws, v_ws, rbias, o_ws);
    proj_kernel<<<BT * LSEQ, 256, 0, stream>>>(o_ws, wp_h, b_proj, out);
}
